// BertMoELayer_31559419691536
// MI455X (gfx1250) — matrix operand order repaired and checked by reading, not compiled, not run on hardware
//
#include <hip/hip_runtime.h>
#include <hip/hip_bf16.h>
#include <math.h>

#define B_ 4
#define S_ 2048
#define H_ 768
#define F_ 3072
#define E_ 8
#define T_ (B_ * S_) /* 8192 tokens */

typedef __bf16 bf16;
typedef bf16 v16bf __attribute__((ext_vector_type(16)));
typedef bf16 v8bf  __attribute__((ext_vector_type(8)));
typedef float v8f  __attribute__((ext_vector_type(8)));

// ---------------------------------------------------------------------------
// WMMA fragment helpers (wave32; this code's contiguous K order, same for A and B)
// A 16x32 bf16: lane<16 -> M=lane, K = 0..15; lane>=16 -> M=lane-16,
// K = 16..31. Two 16B LDS loads per fragment.
// ---------------------------------------------------------------------------
static __device__ __forceinline__ v16bf load_a_frag(const bf16* base, int lda,
                                                    int k0, int lane) {
  int m  = lane & 15;
  int kb = (lane >> 4) << 4;
  const bf16* p = base + m * lda + k0 + kb;
  v8bf lo = *(const v8bf*)(p);
  v8bf hi = *(const v8bf*)(p + 8);
  v16bf r;
#pragma unroll
  for (int j = 0; j < 8; ++j) { r[j] = lo[j]; r[j + 8] = hi[j]; }
  return r;
}

// B 32x16 bf16: lane<16 -> N=lane, K=0..15; lane>=16 -> N=lane-16, K=16..31.
// B[k][n] = W[n-th row][k] for row-major weights contiguous along K:
// one 32B global load per lane.
static __device__ __forceinline__ v16bf load_b_frag(const bf16* base, int ldb,
                                                    int k0, int lane) {
  int n  = lane & 15;
  int kb = (lane >> 4) << 4;
  return *(const v16bf*)(base + (size_t)n * ldb + k0 + kb);
}

static __device__ __forceinline__ v8f wmma_bf16(v16bf a, v16bf b, v8f c) {
  return __builtin_amdgcn_wmma_f32_16x16x32_bf16(false, a, false, b,
                                                 (short)0, c, false, false);
}

// ---------------------------------------------------------------------------
// Kernel 0: zero per-expert counters
// ---------------------------------------------------------------------------
__global__ void init_counts_kernel(int* __restrict__ counts) {
  if (threadIdx.x < E_) counts[threadIdx.x] = 0;
}

// ---------------------------------------------------------------------------
// Kernel 1: fp32 -> bf16 weight conversion (grid-stride)
// ---------------------------------------------------------------------------
__global__ void cvt_f32_bf16_kernel(const float* __restrict__ src,
                                    bf16* __restrict__ dst, int n) {
  int i = blockIdx.x * blockDim.x + threadIdx.x;
  int stride = gridDim.x * blockDim.x;
  for (; i < n; i += stride) dst[i] = (bf16)src[i];
}

// ---------------------------------------------------------------------------
// Kernel 2: router — logits, top-2, softmax over top-2, scatter slot ids.
// Slot id encodes token and rank: (t<<1)|k. Per-expert region capacity = T_.
// ---------------------------------------------------------------------------
__global__ __launch_bounds__(256) void router_kernel(
    const float* __restrict__ x, const float* __restrict__ Wr,
    float* __restrict__ wbuf, int* __restrict__ counts, int* __restrict__ lists) {
  __shared__ float wr[E_ * H_];
  for (int i = threadIdx.x; i < E_ * H_; i += 256) wr[i] = Wr[i];
  __syncthreads();

  int t = blockIdx.x * 256 + threadIdx.x;
  if (t >= T_) return;
  const float* xr = x + (size_t)t * H_;

  float acc[E_];
#pragma unroll
  for (int e = 0; e < E_; ++e) acc[e] = 0.f;
  for (int k = 0; k < H_; ++k) {
    float xv = xr[k];
#pragma unroll
    for (int e = 0; e < E_; ++e) acc[e] += xv * wr[e * H_ + k];
  }

  int i0 = 0;
#pragma unroll
  for (int e = 1; e < E_; ++e) if (acc[e] > acc[i0]) i0 = e;
  int i1 = (i0 == 0) ? 1 : 0;
#pragma unroll
  for (int e = 0; e < E_; ++e)
    if (e != i0 && e != i1 && acc[e] > acc[i1]) i1 = e;

  float ex = expf(acc[i1] - acc[i0]);  // acc[i1] <= acc[i0]
  float denom = 1.f + ex;
  wbuf[2 * t]     = 1.f / denom;
  wbuf[2 * t + 1] = ex / denom;

  int p0 = atomicAdd(&counts[i0], 1);
  lists[i0 * T_ + p0] = (t << 1);
  int p1 = atomicAdd(&counts[i1], 1);
  lists[i1 * T_ + p1] = (t << 1) | 1;
}

// ---------------------------------------------------------------------------
// Kernel 3: fused gathered expert FFN.
// Block = 256 threads (8 wave32 = 2 M-groups x 4 N-waves), 32 token-slots of
// one expert. Per 64-wide F chunk:
//   GEMM1: wave (g,w) computes 16x16 h tile (rows of M-group g, cols f0+16w,
//          K=768, 24 WMMA) -> bias + erf-GELU -> bf16 LDS tile.
//   GEMM2: wave (g,w) accumulates 12 WMMA tiles covering H cols [192w,192w+192)
//          for M-group g (K=64).
// The two M-groups issue identical B-fragment addresses -> WGP$ hits, halving
// L2 weight traffic per token vs a 16-token tile. Per-wave accumulator
// footprint stays at 12 x v8f = 96 VGPRs (no spills).
// ---------------------------------------------------------------------------
#define XS_LD 776   /* 768 + 8 pad: kills LDS bank conflicts, keeps 16B align */
#define HS_LD 72    /* 64 + 8 pad */
#define MT_ 32      /* tokens per block */

__global__ __launch_bounds__(256) void moe_ffn_kernel(
    const float* __restrict__ x,
    const bf16* __restrict__ Wi, const float* __restrict__ bi,
    const bf16* __restrict__ Wo, const float* __restrict__ bo,
    const int* __restrict__ counts, const int* __restrict__ lists,
    float* __restrict__ out_buf) {
  int e = blockIdx.y;
  int cnt = counts[e];
  int mtile = blockIdx.x;
  if (mtile * MT_ >= cnt) return;  // uniform early exit (before any barrier)

  __shared__ __align__(16) bf16 xs[MT_ * XS_LD];
  __shared__ __align__(16) bf16 hs[MT_ * HS_LD];
  __shared__ int slot_s[MT_];

  int tid  = threadIdx.x;
  int lane = tid & 31;
  int wave = tid >> 5;
  int g = wave >> 2;   // M-group: 0 or 1 (rows g*16 .. g*16+15)
  int w = wave & 3;    // N-wave within group

  if (tid < MT_) {
    int gm = mtile * MT_ + tid;
    slot_s[tid] = (gm < cnt) ? lists[e * T_ + gm] : -1;
  }
  __syncthreads();

  // Stage gathered x rows as bf16 (zero-pad invalid tail rows).
  for (int idx = tid; idx < MT_ * H_; idx += 256) {
    int m = idx / H_;
    int k = idx - m * H_;
    int v = slot_s[m];
    float val = (v >= 0) ? x[(size_t)(v >> 1) * H_ + k] : 0.f;
    xs[m * XS_LD + k] = (bf16)val;
  }
  __syncthreads();

  const bf16* WiE  = Wi + (size_t)e * F_ * H_;
  const bf16* WoE  = Wo + (size_t)e * H_ * F_;
  const float* biE = bi + (size_t)e * F_;
  const float* boE = bo + (size_t)e * H_;

  const bf16* xsg = xs + g * 16 * XS_LD;  // this wave's M-group x rows
  const bf16* hsg = hs + g * 16 * HS_LD;  // this wave's M-group h rows

  const v8f zero = {0.f, 0.f, 0.f, 0.f, 0.f, 0.f, 0.f, 0.f};
  v8f acc[12];
#pragma unroll
  for (int i = 0; i < 12; ++i) acc[i] = zero;

  int n_l  = lane & 15;
  int mo_l = (lane >> 4) << 3;  // 0 or 8: accumulator row base (ISA C layout)

  for (int f0 = 0; f0 < F_; f0 += 64) {
    // ---- GEMM1: 16x16 h tile at columns [f0+16*w, +16) for M-group g ----
    int fw = f0 + w * 16;
    const bf16* bb1 = WiE + (size_t)fw * H_;
    if (f0 + 64 < F_)  // prefetch next chunk's Wi rows into cache
      __builtin_prefetch(WiE + (size_t)(fw + 64) * H_ + (n_l << 5), 0, 1);
    v8f hacc = zero;
#pragma unroll 4
    for (int k0 = 0; k0 < H_; k0 += 32) {
      v16bf a = load_a_frag(xsg, XS_LD, k0, lane);
      v16bf b = load_b_frag(bb1, H_, k0, lane);
      hacc = wmma_bf16(a, b, hacc);
    }

    // bias + exact (erf) GELU, store bf16 tile to LDS
    float bv = biE[fw + n_l];
    __syncthreads();  // prior chunk's hs readers are done
#pragma unroll
    for (int r = 0; r < 8; ++r) {
      float hv = hacc[r] + bv;
      hv = 0.5f * hv * (1.f + erff(hv * 0.70710678118654752f));
      hs[(g * 16 + mo_l + r) * HS_LD + w * 16 + n_l] = (bf16)hv;
    }
    __syncthreads();

    // ---- GEMM2: acc[nt] += h_chunk(g) @ Wo[:, f0:f0+64]^T  (K = 64) ----
#pragma unroll
    for (int k0 = 0; k0 < 64; k0 += 32) {
      v16bf a = load_a_frag(hsg, HS_LD, k0, lane);
#pragma unroll
      for (int nt = 0; nt < 12; ++nt) {
        const bf16* bb2 = WoE + (size_t)(w * 192 + nt * 16) * F_ + f0;
        v16bf b = load_b_frag(bb2, F_, k0, lane);
        acc[nt] = wmma_bf16(a, b, acc[nt]);
      }
    }
  }

  // Epilogue: + bo, masked scatter to per-slot output rows.
#pragma unroll
  for (int nt = 0; nt < 12; ++nt) {
    int hcol = w * 192 + nt * 16 + n_l;
    float bov = boE[hcol];
#pragma unroll
    for (int r = 0; r < 8; ++r) {
      int v = slot_s[g * 16 + mo_l + r];
      if (v >= 0) out_buf[(size_t)v * H_ + hcol] = acc[nt][r] + bov;
    }
  }
}

// ---------------------------------------------------------------------------
// Kernel 4: combine top-2 expert rows with routing weights.
// ---------------------------------------------------------------------------
__global__ __launch_bounds__(256) void combine_kernel(
    const float* __restrict__ out_buf, const float* __restrict__ wbuf,
    float* __restrict__ out) {
  int i = blockIdx.x * 256 + threadIdx.x;
  if (i >= T_ * H_) return;
  int t = i / H_;
  int h = i - t * H_;
  out[i] = wbuf[2 * t]     * out_buf[(size_t)(2 * t) * H_ + h] +
           wbuf[2 * t + 1] * out_buf[(size_t)(2 * t + 1) * H_ + h];
}

// ---------------------------------------------------------------------------
extern "C" void kernel_launch(void* const* d_in, const int* in_sizes, int n_in,
                              void* d_out, int out_size, void* d_ws,
                              size_t ws_size, hipStream_t stream) {
  (void)in_sizes; (void)n_in; (void)out_size; (void)ws_size;
  const float* x  = (const float*)d_in[0];
  const float* Wr = (const float*)d_in[1];
  const float* Wi = (const float*)d_in[2];
  const float* bi = (const float*)d_in[3];
  const float* Wo = (const float*)d_in[4];
  const float* bo = (const float*)d_in[5];
  float* out = (float*)d_out;

  char* ws = (char*)d_ws;
  size_t off = 0;
  auto alloc = [&](size_t bytes) {
    off = (off + 255) & ~(size_t)255;
    char* p = ws + off;
    off += bytes;
    return p;
  };
  int*   counts  = (int*)alloc(E_ * sizeof(int));
  int*   lists   = (int*)alloc((size_t)E_ * T_ * sizeof(int));
  float* wbuf    = (float*)alloc((size_t)2 * T_ * sizeof(float));
  float* out_buf = (float*)alloc((size_t)2 * T_ * H_ * sizeof(float));
  bf16*  Wi_b    = (bf16*)alloc((size_t)E_ * F_ * H_ * sizeof(bf16));
  bf16*  Wo_b    = (bf16*)alloc((size_t)E_ * H_ * F_ * sizeof(bf16));

  const int nW = E_ * F_ * H_;

  hipLaunchKernelGGL(init_counts_kernel, dim3(1), dim3(32), 0, stream, counts);
  hipLaunchKernelGGL(cvt_f32_bf16_kernel, dim3(2048), dim3(256), 0, stream,
                     Wi, Wi_b, nW);
  hipLaunchKernelGGL(cvt_f32_bf16_kernel, dim3(2048), dim3(256), 0, stream,
                     Wo, Wo_b, nW);
  hipLaunchKernelGGL(router_kernel, dim3(T_ / 256), dim3(256), 0, stream,
                     x, Wr, wbuf, counts, lists);
  hipLaunchKernelGGL(moe_ffn_kernel, dim3(T_ / MT_, E_), dim3(256), 0, stream,
                     x, Wi_b, bi, Wo_b, bo, counts, lists, out_buf);
  hipLaunchKernelGGL(combine_kernel, dim3((T_ * H_ + 255) / 256), dim3(256),
                     0, stream, out_buf, wbuf, out);
}
